// MANN_78726750536045
// MI455X (gfx1250) — matrix operand order repaired and checked by reading, not compiled, not run on hardware
//
#include <hip/hip_runtime.h>
#include <math.h>

typedef __bf16 bf16_t;
typedef __attribute__((ext_vector_type(16))) __bf16 v16bf;
typedef __attribute__((ext_vector_type(8)))  __bf16 v8bf;
typedef __attribute__((ext_vector_type(4)))  __bf16 v4bf;
typedef __attribute__((ext_vector_type(8)))  float  v8f;

#define BSZ    1024
#define XMAIN  480
#define XGATE  128
#define HID    512
#define YDIM   512
#define GHID   32
#define KEXP   8
#define OUTW   512   // all expert layers have 512 outputs

// ---------------------------------------------------------------------------
// fp32 -> bf16 conversion, 4 elements per thread
// ---------------------------------------------------------------------------
__global__ void cvt_f32_bf16_x4(const float4* __restrict__ src,
                                v4bf* __restrict__ dst, int n4) {
    int i = blockIdx.x * blockDim.x + threadIdx.x;
    if (i < n4) {
        float4 v = src[i];
        v4bf o;
        o[0] = (bf16_t)v.x; o[1] = (bf16_t)v.y;
        o[2] = (bf16_t)v.z; o[3] = (bf16_t)v.w;
        dst[i] = o;
    }
}

// ---------------------------------------------------------------------------
// Gating network: ELU(x@gw1^T+gb1) -> ELU(@gw2^T+gb2) -> softmax(@gw3^T+gb3)
// ---------------------------------------------------------------------------
__global__ __launch_bounds__(128)
void gate_kernel(const float* __restrict__ xg,
                 const float* __restrict__ gw1, const float* __restrict__ gb1,
                 const float* __restrict__ gw2, const float* __restrict__ gb2,
                 const float* __restrict__ gw3, const float* __restrict__ gb3,
                 float* __restrict__ gates) {
    __shared__ float s_w1[GHID][XGATE];
    __shared__ float s_w2[GHID][GHID];
    __shared__ float s_w3[KEXP][GHID];
    __shared__ float s_b1[GHID], s_b2[GHID], s_b3[KEXP];

    const int tid = threadIdx.x;
    for (int i = tid; i < GHID * XGATE; i += 128) s_w1[i / XGATE][i % XGATE] = gw1[i];
    for (int i = tid; i < GHID * GHID;  i += 128) s_w2[i / GHID][i % GHID]  = gw2[i];
    for (int i = tid; i < KEXP * GHID;  i += 128) s_w3[i / GHID][i % GHID]  = gw3[i];
    if (tid < GHID) { s_b1[tid] = gb1[tid]; s_b2[tid] = gb2[tid]; }
    if (tid < KEXP)   s_b3[tid] = gb3[tid];
    __syncthreads();

    const int row = blockIdx.x * 128 + tid;
    if (row >= BSZ) return;

    float x[XGATE];
    #pragma unroll 8
    for (int i = 0; i < XGATE; ++i) x[i] = xg[row * XGATE + i];

    float h1[GHID];
    for (int o = 0; o < GHID; ++o) {
        float a = s_b1[o];
        #pragma unroll 8
        for (int i = 0; i < XGATE; ++i) a += x[i] * s_w1[o][i];
        h1[o] = a > 0.f ? a : (__expf(a) - 1.f);
    }
    float h2[GHID];
    for (int o = 0; o < GHID; ++o) {
        float a = s_b2[o];
        #pragma unroll
        for (int i = 0; i < GHID; ++i) a += h1[i] * s_w2[o][i];
        h2[o] = a > 0.f ? a : (__expf(a) - 1.f);
    }
    float lg[KEXP];
    float mx = -1e30f;
    for (int o = 0; o < KEXP; ++o) {
        float a = s_b3[o];
        #pragma unroll
        for (int i = 0; i < GHID; ++i) a += h2[i] * s_w3[o][i];
        lg[o] = a;
        mx = fmaxf(mx, a);
    }
    float den = 0.f;
    for (int o = 0; o < KEXP; ++o) { lg[o] = __expf(lg[o] - mx); den += lg[o]; }
    const float inv = 1.f / den;
    for (int o = 0; o < KEXP; ++o) gates[row * KEXP + o] = lg[o] * inv;
}

// ---------------------------------------------------------------------------
// fragment helpers
// ---------------------------------------------------------------------------
__device__ __forceinline__ v16bf combine(v8bf lo, v8bf hi) {
    v16bf r;
    #pragma unroll
    for (int j = 0; j < 8; ++j) { r[j] = lo[j]; r[j + 8] = hi[j]; }
    return r;
}

union F16x16 { v16bf v; unsigned u[8]; };
union GPack  { bf16_t b[2]; unsigned u; };

// packed bf16 multiply by a splatted gate: 8 x v_pk_mul_bf16
__device__ __forceinline__ v16bf scale_bf16x16(v16bf a, unsigned gpack) {
    F16x16 in, out;
    in.v = a;
    #pragma unroll
    for (int j = 0; j < 8; ++j)
        asm("v_pk_mul_bf16 %0, %1, %2" : "=v"(out.u[j]) : "v"(in.u[j]), "v"(gpack));
    return out.v;
}

__device__ __forceinline__ v16bf load_b16(const bf16_t* p) {
    return combine(*(const v8bf*)p, *(const v8bf*)(p + 16));
}

// ---------------------------------------------------------------------------
// One MoE expert layer:
//   out[b,o] = ELU?( sum_k g[b,k] * ( act[b,:] . W[k][o,:] + eb[k][o] ) )
// Gate pre-scaling of the A fragment (one batch row per lane) lets all
// 8 experts and all K-chunks accumulate into a single f32 WMMA accumulator.
// Block: 128 threads = 4 wave32. Tile: 64 rows x 64 cols (wave: 16 x 64).
// Full A tile (64 x IN) staged once in dynamic LDS; B fragments stream from
// L2 through a loop-carried ping-pong pipeline (loads of chunk i+1 issued
// before the WMMAs of chunk i, buffers simultaneously live => no merging).
// ---------------------------------------------------------------------------
template <int IN, bool ELU, bool OBF16>
__global__ __launch_bounds__(128, 1)
void moe_layer(const bf16_t* __restrict__ act,   // [B][IN]  bf16
               const bf16_t* __restrict__ W,     // [K][OUTW][IN] bf16
               const float*  __restrict__ eb,    // [K][OUTW]
               const float*  __restrict__ gates, // [B][K]
               void* __restrict__ outp) {
    constexpr int NIC = IN / 32;                 // # of K=32 chunks
    constexpr int LDA = IN + 8;                  // padded LDS row stride (halves)
    extern __shared__ __align__(16) char smem[];
    bf16_t* As = (bf16_t*)smem;                  // [64][LDA]
    float*  Gs = (float*)(smem + (size_t)64 * LDA * sizeof(bf16_t)); // [64][8]

    const int tid  = threadIdx.x;
    const int lane = tid & 31;
    const int wv   = tid >> 5;     // wave 0..3 -> 16-row slice
    const int half = lane >> 4;    // 0|1
    const int r    = lane & 15;
    const int bm   = blockIdx.x * 64;
    const int bn   = blockIdx.y * 64;

    // stage gates (rows bm..bm+63 contiguous: 512 floats)
    for (int i = tid; i < 64 * KEXP; i += 128) Gs[i] = gates[(size_t)bm * KEXP + i];

    // stage full A tile: 64 rows x IN cols bf16, 8-half vectors
    constexpr int VPR = IN / 8;
    for (int i = tid; i < 64 * VPR; i += 128) {
        const int rr = i / VPR;
        const int cc = (i % VPR) * 8;
        *(v8bf*)&As[rr * LDA + cc] = *(const v8bf*)&act[(size_t)(bm + rr) * IN + cc];
    }
    __syncthreads();

    v8f acc[4];
    #pragma unroll
    for (int t = 0; t < 4; ++t)
        #pragma unroll
        for (int j = 0; j < 8; ++j) acc[t][j] = 0.f;

    const int arow = wv * 16 + r;          // local A row for this lane
    const bf16_t* a_lds = &As[arow * LDA + half * 8]; // lane's A row (+K offset)

    #pragma unroll 1
    for (int k = 0; k < KEXP; ++k) {
        const float g = Gs[arow * KEXP + k];
        GPack gp; gp.b[0] = (bf16_t)g; gp.b[1] = (bf16_t)g;
        const unsigned gpack = gp.u;

        const bf16_t* wrow[4];
        #pragma unroll
        for (int t = 0; t < 4; ++t)
            wrow[t] = W + ((size_t)k * OUTW + (bn + t * 16 + r)) * IN + half * 8;

        // loop-carried ping-pong: two chunks per iteration, no buffer copies
        v16bf bA[4], bB[4];
        v16bf aA, aB;
        #pragma unroll
        for (int t = 0; t < 4; ++t) bA[t] = load_b16(wrow[t]);
        aA = combine(*(const v8bf*)(a_lds), *(const v8bf*)(a_lds + 16));

        int ic = 0;
        #pragma unroll 1
        for (; ic + 2 <= NIC; ic += 2) {
            // prefetch chunk ic+1 into the B buffers
            {
                const int off = (ic + 1) * 32;
                #pragma unroll
                for (int t = 0; t < 4; ++t) bB[t] = load_b16(wrow[t] + off);
                aB = combine(*(const v8bf*)(a_lds + off),
                             *(const v8bf*)(a_lds + off + 16));
            }
            // compute chunk ic (bA/aA)
            {
                v16bf ab = scale_bf16x16(aA, gpack);
                #pragma unroll
                for (int t = 0; t < 4; ++t)
                    acc[t] = __builtin_amdgcn_wmma_f32_16x16x32_bf16(
                        false, ab, false, bA[t], (short)0, acc[t], false, false);
            }
            // prefetch chunk ic+2 into the A buffers (for next iteration / tail)
            if (ic + 2 < NIC) {
                const int off = (ic + 2) * 32;
                #pragma unroll
                for (int t = 0; t < 4; ++t) bA[t] = load_b16(wrow[t] + off);
                aA = combine(*(const v8bf*)(a_lds + off),
                             *(const v8bf*)(a_lds + off + 16));
            }
            // compute chunk ic+1 (bB/aB)
            {
                v16bf ab = scale_bf16x16(aB, gpack);
                #pragma unroll
                for (int t = 0; t < 4; ++t)
                    acc[t] = __builtin_amdgcn_wmma_f32_16x16x32_bf16(
                        false, ab, false, bB[t], (short)0, acc[t], false, false);
            }
        }
        if (NIC & 1) {  // tail chunk already resides in bA/aA
            v16bf ab = scale_bf16x16(aA, gpack);
            #pragma unroll
            for (int t = 0; t < 4; ++t)
                acc[t] = __builtin_amdgcn_wmma_f32_16x16x32_bf16(
                    false, ab, false, bA[t], (short)0, acc[t], false, false);
        }
    }

    // epilogue: gate-mixed bias, optional ELU, store
    #pragma unroll
    for (int t = 0; t < 4; ++t) {
        const int o = bn + t * 16 + r;
        float ebv[KEXP];
        #pragma unroll
        for (int k = 0; k < KEXP; ++k) ebv[k] = eb[k * OUTW + o];
        #pragma unroll
        for (int v = 0; v < 8; ++v) {
            const int mloc = wv * 16 + v + half * 8;   // C layout: row = v + 8*half
            float val = acc[t][v];
            #pragma unroll
            for (int k = 0; k < KEXP; ++k) val += Gs[mloc * KEXP + k] * ebv[k];
            if (ELU) val = val > 0.f ? val : (__expf(val) - 1.f);
            const size_t idx = (size_t)(bm + mloc) * OUTW + o;
            if (OBF16) ((bf16_t*)outp)[idx] = (bf16_t)val;
            else       ((float*)outp)[idx]  = val;
        }
    }
}

// ---------------------------------------------------------------------------
extern "C" void kernel_launch(void* const* d_in, const int* in_sizes, int n_in,
                              void* d_out, int out_size, void* d_ws, size_t ws_size,
                              hipStream_t stream) {
    const float* x_main = (const float*)d_in[0];
    const float* x_gate = (const float*)d_in[1];
    const float* gw1 = (const float*)d_in[2];
    const float* gb1 = (const float*)d_in[3];
    const float* gw2 = (const float*)d_in[4];
    const float* gb2 = (const float*)d_in[5];
    const float* gw3 = (const float*)d_in[6];
    const float* gb3 = (const float*)d_in[7];
    const float* ew1 = (const float*)d_in[8];
    const float* eb1 = (const float*)d_in[9];
    const float* ew2 = (const float*)d_in[10];
    const float* eb2 = (const float*)d_in[11];
    const float* ew3 = (const float*)d_in[12];
    const float* eb3 = (const float*)d_in[13];

    // workspace carve-up (all sizes 256B-aligned)
    char* ws = (char*)d_ws;
    float*  gates = (float*)ws;  ws += (size_t)BSZ * KEXP * 4;           //  32 KB
    bf16_t* actX  = (bf16_t*)ws; ws += (size_t)BSZ * XMAIN * 2;          // 960 KB
    bf16_t* act1  = (bf16_t*)ws; ws += (size_t)BSZ * HID * 2;            //   1 MB
    bf16_t* act2  = (bf16_t*)ws; ws += (size_t)BSZ * HID * 2;            //   1 MB
    bf16_t* w1b   = (bf16_t*)ws; ws += (size_t)KEXP * HID * XMAIN * 2;   // 3.9 MB
    bf16_t* w2b   = (bf16_t*)ws; ws += (size_t)KEXP * HID * HID * 2;     // 4.2 MB
    bf16_t* w3b   = (bf16_t*)ws; ws += (size_t)KEXP * HID * HID * 2;     // 4.2 MB
    (void)ws_size; (void)in_sizes; (void)n_in; (void)out_size;

    // fp32 -> bf16 conversions
    {
        int n4;
        n4 = BSZ * XMAIN / 4;
        cvt_f32_bf16_x4<<<(n4 + 255) / 256, 256, 0, stream>>>(
            (const float4*)x_main, (v4bf*)actX, n4);
        n4 = KEXP * HID * XMAIN / 4;
        cvt_f32_bf16_x4<<<(n4 + 255) / 256, 256, 0, stream>>>(
            (const float4*)ew1, (v4bf*)w1b, n4);
        n4 = KEXP * HID * HID / 4;
        cvt_f32_bf16_x4<<<(n4 + 255) / 256, 256, 0, stream>>>(
            (const float4*)ew2, (v4bf*)w2b, n4);
        cvt_f32_bf16_x4<<<(n4 + 255) / 256, 256, 0, stream>>>(
            (const float4*)ew3, (v4bf*)w3b, n4);
    }

    // gating network
    gate_kernel<<<BSZ / 128, 128, 0, stream>>>(x_gate, gw1, gb1, gw2, gb2,
                                               gw3, gb3, gates);

    // three expert layers (dynamic LDS: 64 x (IN+8) bf16 + 64x8 f32 gates)
    dim3 grid(BSZ / 64, OUTW / 64);  // 16 x 8
    const size_t sh1 = (size_t)64 * (XMAIN + 8) * 2 + 64 * KEXP * 4;
    const size_t sh2 = (size_t)64 * (HID   + 8) * 2 + 64 * KEXP * 4;
    moe_layer<XMAIN, true,  true ><<<grid, 128, sh1, stream>>>(
        actX, w1b, eb1, gates, (void*)act1);
    moe_layer<HID,   true,  true ><<<grid, 128, sh2, stream>>>(
        act1, w2b, eb2, gates, (void*)act2);
    moe_layer<HID,   false, false><<<grid, 128, sh2, stream>>>(
        act2, w3b, eb3, gates, d_out);
}
